// LSTMModel_9216999817406
// MI455X (gfx1250) — matrix operand order repaired and checked by reading, not compiled, not run on hardware
//
#include <hip/hip_runtime.h>
#include <hip/hip_fp16.h>

typedef __attribute__((ext_vector_type(16))) _Float16 v16h;
typedef __attribute__((ext_vector_type(8)))  _Float16 v8h;
typedef __attribute__((ext_vector_type(8)))  float    v8f;
typedef __attribute__((ext_vector_type(4)))  int      v4i;

#define HDIM  128
#define G4    512      // 4*H
#define BSZ   64
#define TLEN  256
#define VOCAB 10000
#define EDIM  100

#if defined(__AMDGCN__) && __has_builtin(__builtin_amdgcn_global_load_async_to_lds_b128) && __has_builtin(__builtin_amdgcn_s_wait_asynccnt)
#define HAVE_ASYNC_LDS 1
#else
#define HAVE_ASYNC_LDS 0
#endif

// Fast activations on the scan's sequential critical path: CDNA5 has a
// hardware TANH trans op; sigmoid(x) = 0.5*tanh(0.5x)+0.5 makes every gate
// activation a single TRANS instruction instead of exp + IEEE-divide chains.
__device__ __forceinline__ float fast_tanh(float x) {
#if defined(__AMDGCN__) && __has_builtin(__builtin_amdgcn_tanhf)
  return __builtin_amdgcn_tanhf(x);
#elif defined(__AMDGCN__) && __has_builtin(__builtin_amdgcn_tanh_f32)
  return __builtin_amdgcn_tanh_f32(x);
#else
  return tanhf(x);
#endif
}
__device__ __forceinline__ float fast_sigmoid(float x) {
  return 0.5f * fast_tanh(0.5f * x) + 0.5f;
}

// ---------------------------------------------------------------------------
// Kernel 1: embedding gather + input projection (+ both biases)
//   xproj[t][b][g] = sum_e emb[x[b,t]][e] * W_ih[g][e] + b_ih[g] + b_hh[g]
// K=100 is WMMA-hostile; 1.7 GFLOP total -> VALU float4 dot products.
// ---------------------------------------------------------------------------
__global__ __launch_bounds__(512)
void embed_proj_kernel(const int* __restrict__ x,
                       const float* __restrict__ emb,
                       const float* __restrict__ W_ih,
                       const float* __restrict__ b_ih,
                       const float* __restrict__ b_hh,
                       float* __restrict__ xproj) {
  __shared__ float e[EDIM];
  const int bt = blockIdx.x;          // b*TLEN + t
  const int b = bt >> 8;
  const int t = bt & 255;
  const int tok = x[bt];
  for (int k = threadIdx.x; k < EDIM; k += blockDim.x)
    e[k] = emb[(size_t)tok * EDIM + k];
  __syncthreads();

  const int g = threadIdx.x;          // 512 threads -> one gate output each
  const float4* w4 = (const float4*)(W_ih + (size_t)g * EDIM);
  const float4* e4 = (const float4*)e;
  float acc = b_ih[g] + b_hh[g];
#pragma unroll
  for (int k = 0; k < EDIM / 4; ++k) {
    float4 w = w4[k], ev = e4[k];
    acc = fmaf(w.x, ev.x, acc);
    acc = fmaf(w.y, ev.y, acc);
    acc = fmaf(w.z, ev.z, acc);
    acc = fmaf(w.w, ev.w, acc);
  }
  xproj[((size_t)t * BSZ + b) * G4 + g] = acc;
}

// ---------------------------------------------------------------------------
// Kernel 1b: W_fc f32 -> f16 (once), so the FC kernel's LDS staging becomes a
// pure copy eligible for GLOBAL_LOAD_ASYNC_TO_LDS.
// ---------------------------------------------------------------------------
__global__ __launch_bounds__(256)
void wfc_to_f16_kernel(const float* __restrict__ Wfc, _Float16* __restrict__ out) {
  const size_t i = (size_t)blockIdx.x * 256 + threadIdx.x;
  if (i < (size_t)VOCAB * HDIM) out[i] = (_Float16)Wfc[i];
}

// ---------------------------------------------------------------------------
// Kernel 2: LSTM recurrence, one workgroup (32 waves), LDS-resident W_hh.
// Per step: gates_pre = h @ W_hh^T  (64x512, K=128); each of the 32 waves
// owns one (16-row, 16-h-col) tile and computes all 4 gate tiles for it, so
// i/f/g/o for a cell element live in one lane's registers and c never leaves
// VGPRs across all 256 steps.
// ---------------------------------------------------------------------------
__global__ __launch_bounds__(1024)
void lstm_scan_kernel(const float* __restrict__ xproj,   // [T][B][4H] f32
                      const float* __restrict__ W_hh,    // [4H][H] f32
                      _Float16* __restrict__ hs_out,     // [T][B][H] f16
                      float* __restrict__ h_final,       // [B][H]
                      float* __restrict__ c_final) {     // [B][H]
  __shared__ _Float16 Wsh[G4][HDIM + 8];   // B-operand: row n, contiguous k
  __shared__ _Float16 hsh[BSZ][HDIM + 8];  // A-operand: row m, contiguous k

  const int tid = threadIdx.x;

  // Stage W_hh f32 -> f16 LDS; init h = 0.
  for (int i = tid; i < G4 * HDIM; i += 1024) {
    const int n = i >> 7, k = i & (HDIM - 1);
    Wsh[n][k] = (_Float16)W_hh[i];
  }
  for (int i = tid; i < BSZ * HDIM; i += 1024) {
    const int m = i >> 7, k = i & (HDIM - 1);
    hsh[m][k] = (_Float16)0.f;
  }
  __syncthreads();

  const int wave   = tid >> 5;
  const int lane   = tid & 31;
  const int lr     = lane & 15;
  const bool hiH   = lane >= 16;
  const int mt     = wave >> 3;          // 0..3  -> rows m0..m0+15
  const int jt     = wave & 7;           // 0..7  -> h-cols jt*16..+15
  const int m0     = mt << 4;
  const int col    = (jt << 4) + lr;     // this lane's h column (D N-index)

  float creg[8];
#pragma unroll
  for (int r = 0; r < 8; ++r) creg[r] = 0.f;

  for (int t = 0; t < TLEN; ++t) {
    const float* xp = xproj + (size_t)t * BSZ * G4;
    // Warm the cache for this step's x_proj slice while the GEMM runs:
    // lanes 0..15 cover rows m0..m0+15; one prefetch per gate segment.
#pragma unroll
    for (int gate = 0; gate < 4; ++gate)
      __builtin_prefetch(xp + (size_t)(m0 + lr) * G4 + gate * HDIM + (jt << 4), 0, 3);

    v8f acc[4] = {v8f{}, v8f{}, v8f{}, v8f{}};   // gates i,f,g,o
#pragma unroll
    for (int kc = 0; kc < 4; ++kc) {             // K = 128 in chunks of 32
      const int kb = kc << 5;
      // A fragment (16x32 f16): lanes 0-15 K kb..+7 / kb+16..+23, hi +8
      const int mrow = m0 + lr;
      const int ka = kb + (hiH ? 8 : 0);
      v8h alo = *(const v8h*)&hsh[mrow][ka];
      v8h ahi = *(const v8h*)&hsh[mrow][ka + 16];
      v16h A;
#pragma unroll
      for (int i = 0; i < 8; ++i) { A[i] = alo[i]; A[i + 8] = ahi[i]; }
      const int kbB = kb + (hiH ? 8 : 0);
#pragma unroll
      for (int gate = 0; gate < 4; ++gate) {
        const int n = gate * HDIM + (jt << 4) + lr;   // B column
        v8h b0 = *(const v8h*)&Wsh[n][kbB];
        v8h b1 = *(const v8h*)&Wsh[n][kbB + 16];
        v16h Bf;
#pragma unroll
        for (int i = 0; i < 8; ++i) { Bf[i] = b0[i]; Bf[i + 8] = b1[i]; }
        acc[gate] = __builtin_amdgcn_wmma_f32_16x16x32_f16(
            false, A, false, Bf, (short)0, acc[gate], false, false);
      }
    }
    __syncthreads();   // all waves done reading hsh(t-1)

    // Elementwise cell update: this lane owns rows m0+r(+8), column `col`.
    const bool last = (t == TLEN - 1);
#pragma unroll
    for (int r = 0; r < 8; ++r) {
      const int m = m0 + r + (hiH ? 8 : 0);
      const float* xpb = xp + (size_t)m * G4;
      const float gi = fast_sigmoid(acc[0][r] + xpb[col]);
      const float gf = fast_sigmoid(acc[1][r] + xpb[HDIM + col]);
      const float gg = fast_tanh   (acc[2][r] + xpb[2 * HDIM + col]);
      const float go = fast_sigmoid(acc[3][r] + xpb[3 * HDIM + col]);
      const float c = gf * creg[r] + gi * gg;
      const float h = go * fast_tanh(c);
      creg[r] = c;
      hsh[m][col] = (_Float16)h;
      hs_out[((size_t)t * BSZ + m) * HDIM + col] = (_Float16)h;
      if (last) {
        h_final[m * HDIM + col] = h;
        c_final[m * HDIM + col] = c;
      }
    }
    __syncthreads();   // hsh(t) visible before next step's GEMM
  }
}

// ---------------------------------------------------------------------------
// Kernel 3: logits = hs @ W_fc^T + b_fc   ([16384,128]x[128,10000] f16 WMMA)
// Block: 256 threads (8 waves), covers 128 M rows x 64 N cols.
// B tile staged via async global->LDS copy (ASYNCcnt path) when available.
// ---------------------------------------------------------------------------
__global__ __launch_bounds__(256)
void fc_kernel(const _Float16* __restrict__ hs,      // [T][B][H] f16
               const _Float16* __restrict__ wfc16,   // [V][H] f16
               const float* __restrict__ bfc,        // [V]
               float* __restrict__ out) {            // [B*T][V] f32
  __shared__ _Float16 Wsh[64][HDIM + 8];
  const int n0  = blockIdx.x << 6;
  const int tid = threadIdx.x;

  // Stage 64 rows x 128 halfs of W_fc^T operand: 1024 16-byte chunks.
  for (int c = tid; c < 64 * (HDIM / 8); c += 256) {
    const int r  = c >> 4;
    const int k8 = (c & 15) << 3;
    const int n  = n0 + r;
    if (n < VOCAB) {
#if HAVE_ASYNC_LDS
      __builtin_amdgcn_global_load_async_to_lds_b128(
          (__attribute__((address_space(1))) v4i*)(wfc16 + (size_t)n * HDIM + k8),
          (__attribute__((address_space(3))) v4i*)&Wsh[r][k8], 0, 0);
#else
      *(v8h*)&Wsh[r][k8] = *(const v8h*)(wfc16 + (size_t)n * HDIM + k8);
#endif
    } else {
      *(v8h*)&Wsh[r][k8] = v8h{};
    }
  }
#if HAVE_ASYNC_LDS
  __builtin_amdgcn_s_wait_asynccnt(0);
#endif
  __syncthreads();

  const int wave = tid >> 5, lane = tid & 31;
  const int lr = lane & 15;
  const bool hiH = lane >= 16;
  const int m0 = (blockIdx.y << 7) + (wave << 4);  // 16-row strip, t-contig
  const int b  = m0 >> 8;
  const int t0 = m0 & 255;
  // A rows: m = b*256 + t  ->  hs[t][b][:]
  const _Float16* arow = hs + ((size_t)(t0 + lr) * BSZ + b) * HDIM;

  v8f acc[4] = {v8f{}, v8f{}, v8f{}, v8f{}};
#pragma unroll
  for (int kc = 0; kc < 4; ++kc) {
    const int kb = kc << 5;
    const int ka = kb + (hiH ? 8 : 0);
    v8h alo = *(const v8h*)(arow + ka);
    v8h ahi = *(const v8h*)(arow + ka + 16);
    v16h A;
#pragma unroll
    for (int i = 0; i < 8; ++i) { A[i] = alo[i]; A[i + 8] = ahi[i]; }
    const int kbB = kb + (hiH ? 8 : 0);
#pragma unroll
    for (int j = 0; j < 4; ++j) {
      if (n0 + (j << 4) < VOCAB) {     // uniform per wave (V % 16 == 0)
        const int nl = (j << 4) + lr;
        v8h b0 = *(const v8h*)&Wsh[nl][kbB];
        v8h b1 = *(const v8h*)&Wsh[nl][kbB + 16];
        v16h Bf;
#pragma unroll
        for (int i = 0; i < 8; ++i) { Bf[i] = b0[i]; Bf[i + 8] = b1[i]; }
        acc[j] = __builtin_amdgcn_wmma_f32_16x16x32_f16(
            false, A, false, Bf, (short)0, acc[j], false, false);
      }
    }
  }
#pragma unroll
  for (int j = 0; j < 4; ++j) {
    const int ncol = n0 + (j << 4) + lr;
    if (ncol < VOCAB) {
      const float bias = bfc[ncol];
#pragma unroll
      for (int r = 0; r < 8; ++r) {
        const int m = m0 + r + (hiH ? 8 : 0);
        out[(size_t)m * VOCAB + ncol] = acc[j][r] + bias;
      }
    }
  }
}

// ---------------------------------------------------------------------------
extern "C" void kernel_launch(void* const* d_in, const int* in_sizes, int n_in,
                              void* d_out, int out_size, void* d_ws, size_t ws_size,
                              hipStream_t stream) {
  const int*   x    = (const int*)  d_in[0];
  const float* emb  = (const float*)d_in[1];
  const float* W_ih = (const float*)d_in[2];
  const float* W_hh = (const float*)d_in[3];
  const float* b_ih = (const float*)d_in[4];
  const float* b_hh = (const float*)d_in[5];
  const float* W_fc = (const float*)d_in[6];
  const float* b_fc = (const float*)d_in[7];

  // Workspace layout: x_proj f32 [T][B][4H] | hs f16 [T][B][H] | W_fc f16 [V][H]
  char* ws = (char*)d_ws;
  float*    xproj = (float*)ws;
  _Float16* hsf16 = (_Float16*)(ws + (size_t)TLEN * BSZ * G4 * sizeof(float));
  _Float16* wfc16 = (_Float16*)(ws + (size_t)TLEN * BSZ * G4 * sizeof(float)
                                   + (size_t)TLEN * BSZ * HDIM * sizeof(_Float16));

  float* logits = (float*)d_out;
  float* h_fin  = logits + (size_t)BSZ * TLEN * VOCAB;
  float* c_fin  = h_fin + BSZ * HDIM;

  embed_proj_kernel<<<BSZ * TLEN, 512, 0, stream>>>(x, emb, W_ih, b_ih, b_hh, xproj);
  wfc_to_f16_kernel<<<((size_t)VOCAB * HDIM + 255) / 256, 256, 0, stream>>>(W_fc, wfc16);
  lstm_scan_kernel<<<1, 1024, 0, stream>>>(xproj, W_hh, hsf16, h_fin, c_fin);
  dim3 grid((VOCAB + 63) / 64, (BSZ * TLEN) / 128);
  fc_kernel<<<grid, 256, 0, stream>>>(hsf16, wfc16, b_fc, logits);
}
